// TransposedAttentionBlock_44719199486632
// MI455X (gfx1250) — matrix operand order repaired and checked by reading, not compiled, not run on hardware
//
#include <hip/hip_runtime.h>
#include <hip/hip_bf16.h>

typedef __attribute__((ext_vector_type(16))) __bf16 v16bf;
typedef __attribute__((ext_vector_type(8)))  float  v8f;
typedef unsigned int v4u __attribute__((ext_vector_type(4)));
typedef int          v8i __attribute__((ext_vector_type(8)));
typedef int          v4i __attribute__((ext_vector_type(4)));

#define Bn 16
#define Cc 384
#define Nn 3136
#define C3 1152
#define LDSA 40   // padded LDS row stride (elements) for A tile (128x32)
#define LDSB 40   // padded LDS row stride (elements) for Bt tile (64x32)

#if defined(__gfx1250__) && __has_builtin(__builtin_amdgcn_tensor_load_to_lds) && \
    __has_builtin(__builtin_amdgcn_s_wait_tensorcnt)
#define HAVE_TDM 1
#else
#define HAVE_TDM 0
#endif

union FragBF { v16bf v; uint4 q[2]; };
union Pack8  { uint4 q; __bf16 e[8]; };

// ---------------------------------------------------------------- helpers ---
__global__ void f32_to_bf16_kernel(const float* __restrict__ in,
                                   __bf16* __restrict__ out, int n) {
  for (int i = blockIdx.x * blockDim.x + threadIdx.x; i < n;
       i += gridDim.x * blockDim.x)
    out[i] = (__bf16)in[i];
}

// BN fold: xn = x*scale + shift, per (b,c) row of N elements, store bf16.
__global__ void bn_fold_kernel(const float* __restrict__ x,
                               const float* __restrict__ gamma,
                               const float* __restrict__ beta,
                               const float* __restrict__ mean,
                               const float* __restrict__ var,
                               __bf16* __restrict__ xn) {
  int bc = blockIdx.x;            // b*C + c
  int c  = bc % Cc;
  float s  = gamma[c] * rsqrtf(var[c] + 1e-5f);
  float sh = beta[c] - mean[c] * s;
  const float* xi = x  + (long)bc * Nn;
  __bf16*      xo = xn + (long)bc * Nn;
  for (int n = threadIdx.x; n < Nn; n += blockDim.x)
    xo[n] = (__bf16)(xi[n] * s + sh);
}

// 1/max(||row||,eps) for q,k rows (o<768) per batch.
__global__ void rnorm_kernel(const float* __restrict__ sumsq,
                             float* __restrict__ rn) {
  int i = blockIdx.x * blockDim.x + threadIdx.x;    // 16*768
  if (i >= Bn * 768) return;
  int b = i / 768, o = i % 768;
  rn[i] = 1.0f / fmaxf(sqrtf(sumsq[b * C3 + o]), 1e-12f);
}

// softmax over rows of 384 (fp32 in, bf16 out); one wave per row.
__global__ __launch_bounds__(256) void softmax_kernel(
    const float* __restrict__ logits, __bf16* __restrict__ attn) {
  int wave = threadIdx.x >> 5, lane = threadIdx.x & 31;
  long row = (long)blockIdx.x * 8 + wave;           // Bn*Cc rows
  const float* in = logits + row * Cc;
  float v[12], m = -1e30f;
  #pragma unroll
  for (int i = 0; i < 12; ++i) { v[i] = in[lane + i * 32]; m = fmaxf(m, v[i]); }
  #pragma unroll
  for (int s = 1; s < 32; s <<= 1) m = fmaxf(m, __shfl_xor(m, s, 32));
  float sum = 0.f;
  #pragma unroll
  for (int i = 0; i < 12; ++i) { v[i] = __expf(v[i] - m); sum += v[i]; }
  #pragma unroll
  for (int s = 1; s < 32; s <<= 1) sum += __shfl_xor(sum, s, 32);
  float inv = 1.0f / sum;
  __bf16* out = attn + row * Cc;
  #pragma unroll
  for (int i = 0; i < 12; ++i) out[lane + i * 32] = (__bf16)(v[i] * inv);
}

// ------------------------------------------------------------ WMMA GEMM -----
// D = A(MxK, row major) * B(KxN, row major), block tile 128x64, 8 waves,
// each wave 32x32 via 2x2 v_wmma_f32_16x16x32_bf16.
// A tile (128x32 bf16, 8KB) is staged global->LDS by the Tensor Data Mover
// (one tensor_load_to_lds per K-step, padded to the LDSA row stride via the
// D# pad fields); B tile is loaded and transposed into LDS by the waves.
// Epilogues:
//   EPI 0: qkv  -> bf16 out, + bias, atomic sumsq for rows < 768
//   EPI 1: logits -> f32 out, * rnq[row]*rnk[col]*temperature
//   EPI 2: attn*v -> bf16 out
//   EPI 3: proj -> f32 out, + bias + residual
template <int EPI>
__global__ __launch_bounds__(256) void gemm_bf16_kernel(
    const __bf16* __restrict__ A,  long aBatch, int lda,
    const __bf16* __restrict__ Bm, long bBatch, int ldb,
    int K,
    float* __restrict__ outF, __bf16* __restrict__ outBF,
    long oBatch, int ldc,
    const float* __restrict__ bias,
    const float* __restrict__ resid, long rBatch,
    float* __restrict__ sumsq,
    const float* __restrict__ rnorm,
    const float* __restrict__ temperature) {
  __shared__ __align__(16) __bf16 Als[128 * LDSA];
  __shared__ __align__(16) __bf16 Bls[64 * LDSB];   // stored transposed [n][k]

  const int lane = threadIdx.x & 31;
  const int wave = threadIdx.x >> 5;
  const int wm   = wave >> 1;          // 0..3 -> 32-row slab
  const int wn   = wave & 1;           // 0..1 -> 32-col slab
  const int h    = lane >> 4;
  const int l15  = lane & 15;

  const int n0 = blockIdx.x * 64;
  const int m0 = blockIdx.y * 128;
  const int batch = blockIdx.z;

  const __bf16* Ab = A  + (long)batch * aBatch;
  const __bf16* Bb = Bm + (long)batch * bBatch;

  v8f acc[2][2];
  #pragma unroll
  for (int i = 0; i < 2; ++i)
    #pragma unroll
    for (int j = 0; j < 2; ++j)
      acc[i][j] = (v8f){0.f, 0.f, 0.f, 0.f, 0.f, 0.f, 0.f, 0.f};

#if HAVE_TDM
  // Loop-invariant D# pieces for the A-tile tensor load.
  const unsigned laddr = (unsigned)(uintptr_t)(&Als[0]);   // LDS byte offset
  v8i g1;
  // d0: data_size=2B (code1)<<16 | pad_enable<<20 | pad_interval=3 (64B)<<22
  //     | pad_amount=3 (16B)<<25
  g1[0] = (int)((1u << 16) | (1u << 20) | (3u << 22) | (3u << 25));
  g1[1] = (int)(32u << 16);    // tensor_dim0[15:0]=32 @bit48
  g1[2] = (int)(128u << 16);   // tensor_dim0 hi=0 | tensor_dim1[15:0]=128
  g1[3] = (int)(32u << 16);    // tensor_dim1 hi=0 | tile_dim0=32 @bit112
  g1[4] = (int)128u;           // tile_dim1=128, tile_dim2=0
  g1[5] = (int)(unsigned)lda;  // tensor_dim0_stride (element units), low 32
  g1[6] = 0;                   // stride hi16 | dim1_stride low (unused, 2D)
  g1[7] = 0;
  const v4i gz  = {0, 0, 0, 0};
  const v8i gz8 = {0, 0, 0, 0, 0, 0, 0, 0};
#endif

  for (int kb = 0; kb < K; kb += 32) {
#if HAVE_TDM
    // ---- A tile via Tensor Data Mover (wave 0 only; EXEC-independent) ----
    if (wave == 0) {
      unsigned long long gaddr =
          (unsigned long long)(uintptr_t)(Ab + (long)m0 * lda + kb);
      v4u g0;
      g0[0] = 1u;                                   // count=1, user mode
      g0[1] = laddr;                                // lds_addr
      g0[2] = (unsigned)gaddr;                      // global_addr[31:0]
      g0[3] = (unsigned)((gaddr >> 32) & 0x01FFFFFFu) | 0x80000000u; // type=2
      // 6-arg form (clang-23 / therock headers): groups 0..3 + extra + cpol
      __builtin_amdgcn_tensor_load_to_lds(g0, g1, gz, gz, gz8, 0);
    }
#else
    // ---- fallback: stage A tile 128x32 (row major, contiguous k) ----
    #pragma unroll
    for (int t = threadIdx.x; t < 512; t += 256) {
      int row = t >> 2, seg = t & 3;
      const uint4* src = reinterpret_cast<const uint4*>(
          Ab + (long)(m0 + row) * lda + kb + seg * 8);
      *reinterpret_cast<uint4*>(&Als[row * LDSA + seg * 8]) = *src;
    }
#endif
    // ---- stage B tile 32x64, transposed into Bls[n][k] ----
    {
      int k = threadIdx.x >> 3, nseg = threadIdx.x & 7;
      Pack8 u;
      u.q = *reinterpret_cast<const uint4*>(
          Bb + (long)(kb + k) * ldb + n0 + nseg * 8);
      #pragma unroll
      for (int i = 0; i < 8; ++i)
        Bls[(nseg * 8 + i) * LDSB + k] = u.e[i];
    }
#if HAVE_TDM
    __builtin_amdgcn_s_wait_tensorcnt(0);   // tensor DMA complete (wave 0)
#endif
    __syncthreads();

    FragBF af[2], bfr[2];
    #pragma unroll
    for (int mi = 0; mi < 2; ++mi) {
      int row = wm * 32 + mi * 16 + l15;
      af[mi].q[0] = *reinterpret_cast<const uint4*>(&Als[row * LDSA + h * 8]);
      af[mi].q[1] = *reinterpret_cast<const uint4*>(&Als[row * LDSA + 16 + h * 8]);
    }
    #pragma unroll
    for (int nj = 0; nj < 2; ++nj) {
      int nr = wn * 32 + nj * 16 + l15;
      bfr[nj].q[0] = *reinterpret_cast<const uint4*>(&Bls[nr * LDSB + h * 8]);
      bfr[nj].q[1] = *reinterpret_cast<const uint4*>(&Bls[nr * LDSB + 16 + h * 8]);
    }
    #pragma unroll
    for (int mi = 0; mi < 2; ++mi)
      #pragma unroll
      for (int nj = 0; nj < 2; ++nj)
        acc[mi][nj] = __builtin_amdgcn_wmma_f32_16x16x32_bf16(
            false, af[mi].v, false, bfr[nj].v, (short)0, acc[mi][nj],
            false, false);
    __syncthreads();
  }

  // --------------------------------------------------------- epilogue ------
  float temp = (EPI == 1) ? temperature[0] : 0.f;
  #pragma unroll
  for (int mi = 0; mi < 2; ++mi) {
    #pragma unroll
    for (int nj = 0; nj < 2; ++nj) {
      v8f a = acc[mi][nj];
      int gmBase = m0 + wm * 32 + mi * 16 + h * 8;
      int gn = n0 + wn * 32 + nj * 16 + l15;
      #pragma unroll
      for (int r = 0; r < 8; ++r) {
        int gm = gmBase + r;
        long oidx = (long)batch * oBatch + (long)gm * ldc + gn;
        if (EPI == 0) {
          float v = a[r] + bias[gm];
          outBF[oidx] = (__bf16)v;
          float sq = v * v;
          sq += __shfl_xor(sq, 1, 32);
          sq += __shfl_xor(sq, 2, 32);
          sq += __shfl_xor(sq, 4, 32);
          sq += __shfl_xor(sq, 8, 32);
          if (l15 == 0 && gm < 768)
            atomicAdd(&sumsq[batch * C3 + gm], sq);
        } else if (EPI == 1) {
          float s = rnorm[batch * 768 + gm] *
                    rnorm[batch * 768 + 384 + gn] * temp;
          outF[oidx] = a[r] * s;
        } else if (EPI == 2) {
          outBF[oidx] = (__bf16)a[r];
        } else {  // EPI == 3
          float v = a[r] + bias[gm] +
                    resid[(long)batch * rBatch + (long)gm * ldc + gn];
          outF[oidx] = v;
        }
      }
    }
  }
}

// -------------------------------------------------------------- launcher ---
extern "C" void kernel_launch(void* const* d_in, const int* in_sizes, int n_in,
                              void* d_out, int out_size, void* d_ws,
                              size_t ws_size, hipStream_t stream) {
  const float* x        = (const float*)d_in[0];
  const float* w_qkv    = (const float*)d_in[1];
  const float* b_qkv    = (const float*)d_in[2];
  const float* w_proj   = (const float*)d_in[3];
  const float* b_proj   = (const float*)d_in[4];
  const float* bn_gamma = (const float*)d_in[5];
  const float* bn_beta  = (const float*)d_in[6];
  const float* bn_mean  = (const float*)d_in[7];
  const float* bn_var   = (const float*)d_in[8];
  const float* temp     = (const float*)d_in[9];
  float* out = (float*)d_out;

  char* ws = (char*)d_ws;
  size_t off = 0;
  auto carve = [&](size_t bytes) -> char* {
    char* p = ws + off;
    off = (off + bytes + 255) & ~(size_t)255;
    return p;
  };
  __bf16* wqkv_bf  = (__bf16*)carve((size_t)C3 * Cc * 2);
  __bf16* wproj_bf = (__bf16*)carve((size_t)Cc * Cc * 2);
  __bf16* xn_bf    = (__bf16*)carve((size_t)Bn * Cc * Nn * 2);
  __bf16* qkv_bf   = (__bf16*)carve((size_t)Bn * C3 * Nn * 2);
  float*  sumsq    = (float*) carve((size_t)Bn * C3 * 4);
  float*  rnorm    = (float*) carve((size_t)Bn * 768 * 4);
  float*  logits   = (float*) carve((size_t)Bn * Cc * Cc * 4);
  __bf16* attn_bf  = (__bf16*)carve((size_t)Bn * Cc * Cc * 2);
  __bf16* fused_bf = (__bf16*)carve((size_t)Bn * Cc * Nn * 2);

  // 1. weight conversion + BN fold to bf16
  f32_to_bf16_kernel<<<512, 256, 0, stream>>>(w_qkv, wqkv_bf, C3 * Cc);
  f32_to_bf16_kernel<<<256, 256, 0, stream>>>(w_proj, wproj_bf, Cc * Cc);
  bn_fold_kernel<<<Bn * Cc, 256, 0, stream>>>(x, bn_gamma, bn_beta, bn_mean,
                                              bn_var, xn_bf);
  (void)hipMemsetAsync(sumsq, 0, (size_t)Bn * C3 * 4, stream);

  // 2. qkv = w_qkv @ xn  (+bias, sumsq atomics)    M=1152 N=3136 K=384
  gemm_bf16_kernel<0><<<dim3(Nn / 64, C3 / 128, Bn), 256, 0, stream>>>(
      wqkv_bf, 0, Cc, xn_bf, (long)Cc * Nn, Nn, Cc,
      nullptr, qkv_bf, (long)C3 * Nn, Nn,
      b_qkv, nullptr, 0, sumsq, nullptr, nullptr);

  // 3. row rnorms for q,k
  rnorm_kernel<<<(Bn * 768 + 255) / 256, 256, 0, stream>>>(sumsq, rnorm);

  // 4. logits = q @ k^T * rnq*rnk*T                M=384 N=384 K=3136
  gemm_bf16_kernel<1><<<dim3(Cc / 64, Cc / 128, Bn), 256, 0, stream>>>(
      qkv_bf, (long)C3 * Nn, Nn,
      qkv_bf + (long)Cc * Nn, (long)C3 * Nn, Nn, Nn,
      logits, nullptr, (long)Cc * Cc, Cc,
      nullptr, nullptr, 0, nullptr, rnorm, temp);

  // 5. softmax rows of 384 -> bf16
  softmax_kernel<<<(Bn * Cc) / 8, 256, 0, stream>>>(logits, attn_bf);

  // 6. fused = attn @ v                            M=384 N=3136 K=384
  gemm_bf16_kernel<2><<<dim3(Nn / 64, Cc / 128, Bn), 256, 0, stream>>>(
      attn_bf, (long)Cc * Cc, Cc,
      qkv_bf + (long)768 * Nn, (long)C3 * Nn, Nn, Cc,
      nullptr, fused_bf, (long)Cc * Nn, Nn,
      nullptr, nullptr, 0, nullptr, nullptr, nullptr);

  // 7. out = w_proj @ fused + b_proj + x           M=384 N=3136 K=384
  gemm_bf16_kernel<3><<<dim3(Nn / 64, Cc / 128, Bn), 256, 0, stream>>>(
      wproj_bf, 0, Cc, fused_bf, (long)Cc * Nn, Nn, Cc,
      out, nullptr, (long)Cc * Nn, Nn,
      b_proj, x, (long)Cc * Nn, nullptr, nullptr, nullptr);
}
